// EGNNLayer_54039278518700
// MI455X (gfx1250) — compile-verified, matrix operand order checked by reading, not run on hardware
//
#include <hip/hip_runtime.h>
#include <stdint.h>

#define H   128
#define GS  32
#define EPB 32
#define NPB 32

typedef __bf16 bf16_t;
typedef bf16_t v16bf __attribute__((ext_vector_type(16)));
typedef float  v8f   __attribute__((ext_vector_type(8)));

__device__ __forceinline__ unsigned short f2bf(float f) {
  unsigned u = __float_as_uint(f);
  unsigned r = (u + 0x7FFFu + ((u >> 16) & 1u)) >> 16;   // round-to-nearest-even
  return (unsigned short)r;
}

__device__ __forceinline__ float silu_f(float x) { return x / (1.0f + __expf(-x)); }

__device__ __forceinline__ v8f zero8() {
  v8f z;
#pragma unroll
  for (int i = 0; i < 8; ++i) z[i] = 0.0f;
  return z;
}

union FragU { uint4 u[2]; v16bf v; };

// A-fragment (16x32 bf16) from LDS, per ISA 16-bit A layout:
// lanes 0-15 : row M=lane,   K = k0+0..7  (dw0-3) and k0+16..23 (dw4-7)
// lanes 16-31: row M=lane-16,K = k0+8..15 (dw0-3) and k0+24..31 (dw4-7)
__device__ __forceinline__ v16bf load_a_lds(const unsigned short* s, int stride,
                                            int m0, int k0, int lane) {
  int row = m0 + (lane & 15);
  int b0  = (lane & 16) ? 8 : 0;
  const uint4* p = (const uint4*)(s + row * stride + k0 + b0);
  FragU t;
  t.u[0] = p[0];   // 16 bytes: K base..base+7
  t.u[1] = p[2];   // +32 bytes: K base+16..base+23
  return t.v;
}

// B-fragment pre-packed by pack_w: per-lane 8 contiguous dwords
__device__ __forceinline__ v16bf load_b_frag(const unsigned* w, int frag, int lane) {
  const uint4* p = (const uint4*)(w + (size_t)frag * 256 + lane * 8);
  FragU t;
  t.u[0] = p[0];
  t.u[1] = p[1];
  return t.v;
}

#define WMMA_BF16(a, b, c) \
  __builtin_amdgcn_wmma_f32_16x16x32_bf16(false, (a), false, (b), (short)0, (c), false, false)

// ---------------------------------------------------------------------------
// Zero workspace accumulators
// ---------------------------------------------------------------------------
__global__ void zero_f(float* p, long n) {
  long i  = (long)blockIdx.x * blockDim.x + threadIdx.x;
  long st = (long)gridDim.x * blockDim.x;
  for (; i < n; i += st) p[i] = 0.0f;
}

// ---------------------------------------------------------------------------
// Pack fp32 weight [K,128] row-major into bf16 WMMA B-fragments.
// Fragment (kt,nt) = 32x16 block; lane l dword d holds W[k, n], W[k+1, n] with
//   k = kt*32 + (l>=16 ? 8:0) + (d>=4 ? 16:0) + 2*(d&3),  n = nt*16 + (l&15)
// ---------------------------------------------------------------------------
__global__ void pack_w(const float* __restrict__ W, int KT, unsigned* __restrict__ out) {
  int id    = blockIdx.x * blockDim.x + threadIdx.x;
  int total = KT * 2048;
  if (id >= total) return;
  int kt   = id >> 11;
  int r1   = id & 2047;
  int ntl  = r1 >> 8;
  int r2   = r1 & 255;
  int lane = r2 >> 3;
  int dd   = r2 & 7;
  int base = ((lane & 16) ? 8 : 0) + ((dd & 4) ? 16 : 0) + 2 * (dd & 3);
  int k    = kt * 32 + base;
  int n    = ntl * 16 + (lane & 15);
  unsigned lo = f2bf(W[(size_t)k * H + n]);
  unsigned hi = f2bf(W[(size_t)(k + 1) * H + n]);
  out[id] = lo | (hi << 16);
}

// ---------------------------------------------------------------------------
// Edge kernel: 32 edges/block, 8 waves. Edge MLP (2 GEMMs) + coord MLP,
// atomic scatter into agg[N,H] and posacc[N,3].
// ---------------------------------------------------------------------------
__global__ __launch_bounds__(256) void egnn_edge_kernel(
    const float* __restrict__ h, const float* __restrict__ pos,
    const int* __restrict__ srcI, const int* __restrict__ dstI,
    const float* __restrict__ rbf,
    const float* __restrict__ be1, const float* __restrict__ be2,
    const float* __restrict__ bc1, const float* __restrict__ wc2,
    const float* __restrict__ bc2,
    const unsigned* __restrict__ pWe1, const unsigned* __restrict__ pWe2,
    const unsigned* __restrict__ pWc1,
    float* __restrict__ agg, float* __restrict__ posacc, long E) {
  __shared__ __align__(16) unsigned short sA[EPB * 288];   // [h_src|h_dst|rbf] bf16
  __shared__ __align__(16) unsigned short sHid[EPB * H];   // silu(GEMM1) bf16
  __shared__ __align__(16) unsigned short sEF[EPB * H];    // edge_feat bf16
  __shared__ float sVec[EPB][3];
  __shared__ float sDist[EPB];
  __shared__ int   sDst[EPB];
  __shared__ float sCW[EPB];

  const int  tid = threadIdx.x;
  const long e0  = (long)blockIdx.x * EPB;

  // ---- stage: gather h[src], h[dst]; compute edge_vec/dist ----
  {
    int  row = tid >> 3, c = tid & 7;
    long e   = e0 + row;
    long ec  = (e < E) ? e : 0;
    int  s   = srcI[ec], d = dstI[ec];
    if (c == 0) {
      float vx = pos[(size_t)d * 3 + 0] - pos[(size_t)s * 3 + 0];
      float vy = pos[(size_t)d * 3 + 1] - pos[(size_t)s * 3 + 1];
      float vz = pos[(size_t)d * 3 + 2] - pos[(size_t)s * 3 + 2];
      float dist = sqrtf(vx * vx + vy * vy + vz * vz);
      sVec[row][0] = vx; sVec[row][1] = vy; sVec[row][2] = vz;
      sDist[row] = dist; sDst[row] = d; sCW[row] = 0.0f;
    }
    const float4* hs = (const float4*)(h + (size_t)s * H + c * 16);
    const float4* hd = (const float4*)(h + (size_t)d * H + c * 16);
    unsigned short* ar = sA + row * 288;
#pragma unroll
    for (int q = 0; q < 4; ++q) {
      float4 a = hs[q], b = hd[q];
      int o = c * 16 + q * 4;
      ar[o + 0] = f2bf(a.x); ar[o + 1] = f2bf(a.y);
      ar[o + 2] = f2bf(a.z); ar[o + 3] = f2bf(a.w);
      ar[128 + o + 0] = f2bf(b.x); ar[128 + o + 1] = f2bf(b.y);
      ar[128 + o + 2] = f2bf(b.z); ar[128 + o + 3] = f2bf(b.w);
    }
  }
  __syncthreads();
  // ---- RBF features (need dist) ----
  {
    int row = tid >> 3, c = tid & 7;
    float dist = sDist[row];
    unsigned short* ar = sA + row * 288;
#pragma unroll
    for (int j = 0; j < 4; ++j) {
      float t = dist - rbf[c * 4 + j];
      ar[256 + c * 4 + j] = f2bf(__expf(-10.0f * t * t));
    }
  }
  __syncthreads();

  const int lane = tid & 31;
  const int nt   = tid >> 5;              // wave id = N-tile
  const int ng   = nt * 16 + (lane & 15); // global output column
  const int mb   = (lane & 16) ? 8 : 0;   // C/D row base for this lane half

  // ---- GEMM1: [32,288] @ W_e1 ----
  v8f acc0 = zero8(), acc1 = zero8();
  for (int kt = 0; kt < 9; ++kt) {
    v16bf b  = load_b_frag(pWe1, kt * 8 + nt, lane);
    v16bf aA = load_a_lds(sA, 288, 0, kt * 32, lane);
    v16bf aB = load_a_lds(sA, 288, 16, kt * 32, lane);
    acc0 = WMMA_BF16(aA, b, acc0);
    acc1 = WMMA_BF16(aB, b, acc1);
  }
  {
    float b1 = be1[ng];
#pragma unroll
    for (int r = 0; r < 8; ++r) {
      sHid[(mb + r) * H + ng]      = f2bf(silu_f(acc0[r] + b1));
      sHid[(16 + mb + r) * H + ng] = f2bf(silu_f(acc1[r] + b1));
    }
  }
  __syncthreads();

  // ---- GEMM2: edge_feat = hid @ W_e2 + b_e2; scatter into agg ----
  acc0 = zero8(); acc1 = zero8();
  for (int kt = 0; kt < 4; ++kt) {
    v16bf b  = load_b_frag(pWe2, kt * 8 + nt, lane);
    v16bf aA = load_a_lds(sHid, H, 0, kt * 32, lane);
    v16bf aB = load_a_lds(sHid, H, 16, kt * 32, lane);
    acc0 = WMMA_BF16(aA, b, acc0);
    acc1 = WMMA_BF16(aB, b, acc1);
  }
  {
    float b2 = be2[ng];
#pragma unroll
    for (int r = 0; r < 8; ++r) {
      float x0 = acc0[r] + b2;
      float x1 = acc1[r] + b2;
      int m0 = mb + r, m1 = 16 + mb + r;
      sEF[m0 * H + ng] = f2bf(x0);
      sEF[m1 * H + ng] = f2bf(x1);
      if (e0 + m0 < E) atomicAdd(&agg[(size_t)sDst[m0] * H + ng], x0);
      if (e0 + m1 < E) atomicAdd(&agg[(size_t)sDst[m1] * H + ng], x1);
    }
  }
  __syncthreads();

  // ---- GEMM3: coord MLP hidden; reduce (.. @ W_c2) across lanes via LDS ----
  acc0 = zero8(); acc1 = zero8();
  for (int kt = 0; kt < 4; ++kt) {
    v16bf b  = load_b_frag(pWc1, kt * 8 + nt, lane);
    v16bf aA = load_a_lds(sEF, H, 0, kt * 32, lane);
    v16bf aB = load_a_lds(sEF, H, 16, kt * 32, lane);
    acc0 = WMMA_BF16(aA, b, acc0);
    acc1 = WMMA_BF16(aB, b, acc1);
  }
  {
    float bc = bc1[ng];
    float w2 = wc2[ng];
#pragma unroll
    for (int r = 0; r < 8; ++r) {
      atomicAdd(&sCW[mb + r],      silu_f(acc0[r] + bc) * w2);
      atomicAdd(&sCW[16 + mb + r], silu_f(acc1[r] + bc) * w2);
    }
  }
  __syncthreads();

  // ---- position update scatter ----
  if (tid < EPB) {
    long e = e0 + tid;
    if (e < E) {
      float cw  = sCW[tid] + bc2[0];
      float inv = 1.0f / (sDist[tid] + 1e-8f);
      int   d   = sDst[tid];
#pragma unroll
      for (int k = 0; k < 3; ++k)
        atomicAdd(&posacc[(size_t)d * 3 + k], cw * sVec[tid][k] * inv);
    }
  }
}

// ---------------------------------------------------------------------------
// Node kernel: 32 nodes/block. Residual node MLP + LayerNorm + pos output.
// ---------------------------------------------------------------------------
__global__ __launch_bounds__(256) void egnn_node_kernel(
    const float* __restrict__ h, const float* __restrict__ pos,
    const float* __restrict__ agg, const float* __restrict__ posacc,
    const float* __restrict__ bn1, const float* __restrict__ bn2,
    const float* __restrict__ bsc,
    const float* __restrict__ lng, const float* __restrict__ lnb,
    const unsigned* __restrict__ pWn1, const unsigned* __restrict__ pWn2,
    const unsigned* __restrict__ pWsc,
    float* __restrict__ outH, float* __restrict__ outP, long NN) {
  __shared__ __align__(16) unsigned short sA[NPB * 256];  // [h | agg] bf16
  __shared__ __align__(16) unsigned short sHid[NPB * H];
  __shared__ __align__(16) float sX[NPB * H];             // pre-LN f32
  __shared__ float sMu[NPB], sRs[NPB];

  const int  tid = threadIdx.x;
  const long n0  = (long)blockIdx.x * NPB;

  {
    int  row = tid >> 3, c = tid & 7;
    long n   = n0 + row;
    long nc  = (n < NN) ? n : 0;
    const float4* ph = (const float4*)(h + (size_t)nc * H + c * 16);
    const float4* pa = (const float4*)(agg + (size_t)nc * H + c * 16);
    unsigned short* ar = sA + row * 256;
#pragma unroll
    for (int q = 0; q < 4; ++q) {
      float4 a = ph[q], b = pa[q];
      int o = c * 16 + q * 4;
      ar[o + 0] = f2bf(a.x); ar[o + 1] = f2bf(a.y);
      ar[o + 2] = f2bf(a.z); ar[o + 3] = f2bf(a.w);
      ar[128 + o + 0] = f2bf(b.x); ar[128 + o + 1] = f2bf(b.y);
      ar[128 + o + 2] = f2bf(b.z); ar[128 + o + 3] = f2bf(b.w);
    }
  }
  __syncthreads();

  const int lane = tid & 31;
  const int nt   = tid >> 5;
  const int ng   = nt * 16 + (lane & 15);
  const int mb   = (lane & 16) ? 8 : 0;

  // ---- GEMM: node_in @ W_n1 and node_in @ W_sc (shared A fragments) ----
  v8f a10 = zero8(), a11 = zero8(), s0 = zero8(), s1 = zero8();
  for (int kt = 0; kt < 8; ++kt) {
    v16bf b1 = load_b_frag(pWn1, kt * 8 + nt, lane);
    v16bf bs = load_b_frag(pWsc, kt * 8 + nt, lane);
    v16bf aA = load_a_lds(sA, 256, 0, kt * 32, lane);
    v16bf aB = load_a_lds(sA, 256, 16, kt * 32, lane);
    a10 = WMMA_BF16(aA, b1, a10);
    a11 = WMMA_BF16(aB, b1, a11);
    s0  = WMMA_BF16(aA, bs, s0);
    s1  = WMMA_BF16(aB, bs, s1);
  }
  {
    float b1 = bn1[ng];
#pragma unroll
    for (int r = 0; r < 8; ++r) {
      sHid[(mb + r) * H + ng]      = f2bf(silu_f(a10[r] + b1));
      sHid[(16 + mb + r) * H + ng] = f2bf(silu_f(a11[r] + b1));
    }
  }
  __syncthreads();

  // ---- GEMM: hid @ W_n2; combine with shortcut ----
  v8f c0 = zero8(), c1 = zero8();
  for (int kt = 0; kt < 4; ++kt) {
    v16bf b  = load_b_frag(pWn2, kt * 8 + nt, lane);
    v16bf aA = load_a_lds(sHid, H, 0, kt * 32, lane);
    v16bf aB = load_a_lds(sHid, H, 16, kt * 32, lane);
    c0 = WMMA_BF16(aA, b, c0);
    c1 = WMMA_BF16(aB, b, c1);
  }
  {
    float b2 = bn2[ng], bv = bsc[ng];
#pragma unroll
    for (int r = 0; r < 8; ++r) {
      sX[(mb + r) * H + ng]      = c0[r] + b2 + s0[r] + bv;
      sX[(16 + mb + r) * H + ng] = c1[r] + b2 + s1[r] + bv;
    }
  }
  __syncthreads();

  // ---- LayerNorm stats (population var, eps=1e-5) ----
  if (tid < NPB) {
    float mu = 0.0f, m2 = 0.0f;
    const float* xr = sX + tid * H;
    for (int i = 0; i < H; ++i) { float x = xr[i]; mu += x; m2 += x * x; }
    mu *= (1.0f / H); m2 *= (1.0f / H);
    sMu[tid] = mu;
    sRs[tid] = rsqrtf(m2 - mu * mu + 1e-5f);
  }
  __syncthreads();

  {
    int  row = tid >> 3, c = tid & 7;
    long n   = n0 + row;
    if (n < NN) {
      float mu = sMu[row], rs = sRs[row];
      const float* xr  = sX + row * H;
      float* orow = outH + (size_t)n * H;
#pragma unroll
      for (int i = 0; i < 16; ++i) {
        int col = c * 16 + i;
        orow[col] = (xr[col] - mu) * rs * lng[col] + lnb[col];
      }
    }
  }
  if (tid < NPB) {
    long n = n0 + tid;
    if (n < NN) {
#pragma unroll
      for (int k = 0; k < 3; ++k)
        outP[(size_t)n * 3 + k] = pos[(size_t)n * 3 + k] + posacc[(size_t)n * 3 + k];
    }
  }
}

// ---------------------------------------------------------------------------
extern "C" void kernel_launch(void* const* d_in, const int* in_sizes, int n_in,
                              void* d_out, int out_size, void* d_ws, size_t ws_size,
                              hipStream_t stream) {
  const float* h    = (const float*)d_in[0];
  const float* pos  = (const float*)d_in[1];
  const int*   eidx = (const int*)d_in[2];
  const float* rbf  = (const float*)d_in[3];
  const float* We1  = (const float*)d_in[4];
  const float* be1  = (const float*)d_in[5];
  const float* We2  = (const float*)d_in[6];
  const float* be2  = (const float*)d_in[7];
  const float* Wn1  = (const float*)d_in[8];
  const float* bn1  = (const float*)d_in[9];
  const float* Wn2  = (const float*)d_in[10];
  const float* bn2  = (const float*)d_in[11];
  const float* Wsc  = (const float*)d_in[12];
  const float* bsc  = (const float*)d_in[13];
  const float* lng  = (const float*)d_in[14];
  const float* lnb  = (const float*)d_in[15];
  const float* Wc1  = (const float*)d_in[16];
  const float* bc1  = (const float*)d_in[17];
  const float* Wc2  = (const float*)d_in[18];
  const float* bc2  = (const float*)d_in[19];

  const long NN = in_sizes[0] / H;       // 50000
  const long E  = in_sizes[2] / 2;       // 800000

  // workspace layout
  float*    agg    = (float*)d_ws;          // NN*H
  float*    posacc = agg + (size_t)NN * H;  // NN*3
  unsigned* wp     = (unsigned*)(posacc + (size_t)NN * 3);
  unsigned* pWe1 = wp;                 // 9 ktiles * 8 nt * 256 dw
  unsigned* pWe2 = wp + 18432;         // 4 * 8 * 256
  unsigned* pWn1 = wp + 26624;         // 8 * 8 * 256
  unsigned* pWn2 = wp + 43008;         // 4 * 8 * 256
  unsigned* pWsc = wp + 51200;         // 8 * 8 * 256
  unsigned* pWc1 = wp + 67584;         // 4 * 8 * 256

  const int* srcI = eidx;
  const int* dstI = eidx + E;

  zero_f<<<2048, 256, 0, stream>>>(agg, (long)NN * H + NN * 3);

  pack_w<<<(9 * 2048 + 255) / 256, 256, 0, stream>>>(We1, 9, pWe1);
  pack_w<<<(4 * 2048 + 255) / 256, 256, 0, stream>>>(We2, 4, pWe2);
  pack_w<<<(8 * 2048 + 255) / 256, 256, 0, stream>>>(Wn1, 8, pWn1);
  pack_w<<<(4 * 2048 + 255) / 256, 256, 0, stream>>>(Wn2, 4, pWn2);
  pack_w<<<(8 * 2048 + 255) / 256, 256, 0, stream>>>(Wsc, 8, pWsc);
  pack_w<<<(4 * 2048 + 255) / 256, 256, 0, stream>>>(Wc1, 4, pWc1);

  int eBlocks = (int)((E + EPB - 1) / EPB);
  egnn_edge_kernel<<<eBlocks, 256, 0, stream>>>(
      h, pos, srcI, dstI, rbf, be1, be2, bc1, Wc2, bc2,
      pWe1, pWe2, pWc1, agg, posacc, E);

  float* outH = (float*)d_out;
  float* outP = outH + (size_t)NN * H;
  int nBlocks = (int)((NN + NPB - 1) / NPB);
  egnn_node_kernel<<<nBlocks, 256, 0, stream>>>(
      h, pos, agg, posacc, bn1, bn2, bsc, lng, lnb,
      pWn1, pWn2, pWsc, outH, outP, NN);
}
